// SingleHeadedAttention_18648747999689
// MI455X (gfx1250) — matrix operand order repaired and checked by reading, not compiled, not run on hardware
//
#include <hip/hip_runtime.h>
#include <hip/hip_bf16.h>

#define SEQ    4096
#define DMODEL 768
#define DHEAD  64
#define NBATCH 2
#define BS     (SEQ * NBATCH)

typedef __attribute__((ext_vector_type(16))) _Float16 v16h;
typedef __attribute__((ext_vector_type(8)))  _Float16 v8h;
typedef __attribute__((ext_vector_type(8)))  float    v8f;

__device__ __forceinline__ v8f wmma_f32_f16(v16h a, v16h b, v8f c) {
  // D = A(16x32 f16) * B(32x16 f16) + C(16x16 f32)
  return __builtin_amdgcn_wmma_f32_16x16x32_f16(false, a, false, b, (short)0, c,
                                                false, false);
}

// ---- Fragment loaders (all-f16 sources -> pure b128 loads) -----------------
// A fragment: 16(M) x 32(K), row-major source with row stride ld elements.
// Lanes 0-15: M=lane, K={0..7,16..23}; lanes 16-31: M=lane-16, K={8..15,24..31}.
__device__ __forceinline__ v16h frag_a_h(const _Float16* p, int ld) {
  const int lane = threadIdx.x & 31;
  const int r    = lane & 15;
  const int hf   = lane >> 4;
  const _Float16* base = p + (size_t)r * ld + hf * 8;
  v8h lo = *(const v8h*)(base);
  v8h hi = *(const v8h*)(base + 16);
  v16h a;
#pragma unroll
  for (int i = 0; i < 8; ++i) { a[i] = lo[i]; a[i + 8] = hi[i]; }
  return a;
}

// B fragment: 32(K) x 16(N). Source R is the N-major operand: R[n][k],
// row stride ld. Lanes 0-15: N=lane, K={0..7,16..23}; lanes 16-31: N=lane-16, K={8..15,24..31}.
__device__ __forceinline__ v16h frag_b_h(const _Float16* p, int ld) {
  const int lane = threadIdx.x & 31;
  const _Float16* base = p + (size_t)(lane & 15) * ld + (lane >> 4) * 8;
  v8h lo = *(const v8h*)(base);
  v8h hi = *(const v8h*)(base + 16);
  v16h b;
#pragma unroll
  for (int i = 0; i < 8; ++i) { b[i] = lo[i]; b[i + 8] = hi[i]; }
  return b;
}

// ---- Kernel 0: streaming f32 -> f16 conversion -----------------------------
__global__ __launch_bounds__(256) void cvt_f16_kernel(
    const float* __restrict__ src, _Float16* __restrict__ dst, int n) {
  const int i = (blockIdx.x * 256 + threadIdx.x) * 8;
  if (i + 8 <= n) {
    const float4 f0 = *(const float4*)(src + i);
    const float4 f1 = *(const float4*)(src + i + 4);
    v8h h;
    h[0] = (_Float16)f0.x; h[1] = (_Float16)f0.y;
    h[2] = (_Float16)f0.z; h[3] = (_Float16)f0.w;
    h[4] = (_Float16)f1.x; h[5] = (_Float16)f1.y;
    h[6] = (_Float16)f1.z; h[7] = (_Float16)f1.w;
    *(v8h*)(dst + i) = h;
  } else {
    for (int j = i; j < n; ++j) dst[j] = (_Float16)src[j];
  }
}

// ---- Kernel 1: fused QKV projection (xh @ W^T x3), all f16 in/out ----------
// grid = (BS/64); block = 128 (4 waves, 16 rows each). One x A-fragment feeds
// all 12 weight B-fragments -> 12 WMMAs per K-step.
// q scaled by 1/sqrt(64); v stored TRANSPOSED (B, DHEAD, SEQ).
__global__ __launch_bounds__(128) void qkv_proj_kernel(
    const _Float16* __restrict__ xh, const _Float16* __restrict__ wq,
    const _Float16* __restrict__ wk, const _Float16* __restrict__ wv,
    _Float16* __restrict__ qb, _Float16* __restrict__ kbuf,
    _Float16* __restrict__ vT) {
  const int wave = threadIdx.x >> 5;
  const int lane = threadIdx.x & 31;
  const int n    = lane & 15;
  const int hf   = lane >> 4;
  const int row0 = blockIdx.x * 64 + wave * 16;

  v8f acc[3][4];
#pragma unroll
  for (int w = 0; w < 3; ++w)
#pragma unroll
    for (int nt = 0; nt < 4; ++nt) acc[w][nt] = (v8f){0, 0, 0, 0, 0, 0, 0, 0};

  const _Float16* ws_[3] = {wq, wk, wv};
  for (int kt = 0; kt < DMODEL; kt += 32) {
    const v16h a = frag_a_h(xh + (size_t)row0 * DMODEL + kt, DMODEL);
#pragma unroll
    for (int w = 0; w < 3; ++w) {
#pragma unroll
      for (int nt = 0; nt < 4; ++nt) {
        v16h bw = frag_b_h(ws_[w] + (size_t)(nt * 16) * DMODEL + kt, DMODEL);
        acc[w][nt] = wmma_f32_f16(a, bw, acc[w][nt]);
      }
    }
  }

#pragma unroll
  for (int nt = 0; nt < 4; ++nt) {
#pragma unroll
    for (int r = 0; r < 8; ++r) {
      const int grow = row0 + r + 8 * hf;
      const int h    = nt * 16 + n;
      const int bidx = grow >> 12;        // / SEQ
      const int s    = grow & (SEQ - 1);  // % SEQ
      qb[(size_t)grow * DHEAD + h]   = (_Float16)(acc[0][nt][r] * 0.125f);
      kbuf[(size_t)grow * DHEAD + h] = (_Float16)acc[1][nt][r];
      vT[((size_t)bidx * DHEAD + h) * SEQ + s] = (_Float16)acc[2][nt][r];
    }
  }
}

// ---- Kernel 2: causal flash attention --------------------------------------
// grid = (SEQ/64, NBATCH); block = 128. One wave owns 16 query rows.
__global__ __launch_bounds__(128) void flash_attn_kernel(
    const _Float16* __restrict__ qb, const _Float16* __restrict__ kbuf,
    const _Float16* __restrict__ vT, _Float16* __restrict__ zb) {
  __shared__ _Float16 Plds[4][16 * 32];  // per-wave 16x32 f16 P staging
  const int wave = threadIdx.x >> 5;
  const int lane = threadIdx.x & 31;
  const int n    = lane & 15;
  const int hf   = lane >> 4;
  const int b    = blockIdx.y;
  const int q0   = blockIdx.x * 64 + wave * 16;

  const _Float16* qp = qb + (size_t)b * SEQ * DHEAD;
  const _Float16* kp = kbuf + (size_t)b * SEQ * DHEAD;
  const _Float16* vp = vT + (size_t)b * DHEAD * SEQ;
  _Float16* pl = &Plds[wave][0];

  const v16h aQ0 = frag_a_h(qp + (size_t)q0 * DHEAD, DHEAD);
  const v16h aQ1 = frag_a_h(qp + (size_t)q0 * DHEAD + 32, DHEAD);

  float mr[8], lr[8], cr[8];
  v8f accO[4];
#pragma unroll
  for (int r = 0; r < 8; ++r) { mr[r] = -1e30f; lr[r] = 0.0f; }
#pragma unroll
  for (int nt = 0; nt < 4; ++nt) accO[nt] = (v8f){0, 0, 0, 0, 0, 0, 0, 0};

  for (int kb = 0; kb < q0 + 16; kb += 32) {  // causal: keys <= q0+15
    v8f s0 = (v8f){0, 0, 0, 0, 0, 0, 0, 0};
    v8f s1 = (v8f){0, 0, 0, 0, 0, 0, 0, 0};
    s0 = wmma_f32_f16(aQ0, frag_b_h(kp + (size_t)kb * DHEAD, DHEAD), s0);
    s0 = wmma_f32_f16(aQ1, frag_b_h(kp + (size_t)kb * DHEAD + 32, DHEAD), s0);
    s1 = wmma_f32_f16(aQ0, frag_b_h(kp + (size_t)(kb + 16) * DHEAD, DHEAD), s1);
    s1 = wmma_f32_f16(aQ1, frag_b_h(kp + (size_t)(kb + 16) * DHEAD + 32, DHEAD), s1);

    const int key0 = kb + n;
    const int key1 = kb + 16 + n;
#pragma unroll
    for (int r = 0; r < 8; ++r) {
      const int row  = r + 8 * hf;
      const int qrow = q0 + row;
      float x0 = (key0 <= qrow) ? s0[r] : -1e30f;
      float x1 = (key1 <= qrow) ? s1[r] : -1e30f;
      // row-max across the 16 lanes holding this row (stays in N group)
      float rm = fmaxf(x0, x1);
      rm = fmaxf(rm, __shfl_xor(rm, 1));
      rm = fmaxf(rm, __shfl_xor(rm, 2));
      rm = fmaxf(rm, __shfl_xor(rm, 4));
      rm = fmaxf(rm, __shfl_xor(rm, 8));
      const float mn   = fmaxf(mr[r], rm);
      const float corr = __expf(mr[r] - mn);
      const float p0   = __expf(x0 - mn);
      const float p1   = __expf(x1 - mn);
      float rs = p0 + p1;
      rs += __shfl_xor(rs, 1);
      rs += __shfl_xor(rs, 2);
      rs += __shfl_xor(rs, 4);
      rs += __shfl_xor(rs, 8);
      lr[r] = lr[r] * corr + rs;
      mr[r] = mn;
      cr[r] = corr;
      // stage P row-major in LDS (C-layout -> memory) for A-layout regather
      pl[row * 32 + n]      = (_Float16)p0;
      pl[row * 32 + 16 + n] = (_Float16)p1;
    }
    asm volatile("s_wait_dscnt 0" ::: "memory");
    const v16h aP = frag_a_h(pl, 32);
#pragma unroll
    for (int nt = 0; nt < 4; ++nt) {
#pragma unroll
      for (int r = 0; r < 8; ++r) accO[nt][r] *= cr[r];
      v16h bV = frag_b_h(vp + (size_t)(nt * 16) * SEQ + kb, SEQ);
      accO[nt] = wmma_f32_f16(aP, bV, accO[nt]);
    }
  }

#pragma unroll
  for (int nt = 0; nt < 4; ++nt) {
#pragma unroll
    for (int r = 0; r < 8; ++r) {
      const int row = r + 8 * hf;
      const float val = accO[nt][r] / lr[r];
      zb[((size_t)b * SEQ + q0 + row) * DHEAD + nt * 16 + n] = (_Float16)val;
    }
  }
}

// ---- Kernel 3: output projection z @ Wo^T, f32 output ----------------------
// grid = (BS/64); block = 128.
__global__ __launch_bounds__(128) void out_proj_kernel(
    const _Float16* __restrict__ zb, const _Float16* __restrict__ woh,
    float* __restrict__ out) {
  const int wave = threadIdx.x >> 5;
  const int lane = threadIdx.x & 31;
  const int n    = lane & 15;
  const int hf   = lane >> 4;
  const int row0 = blockIdx.x * 64 + wave * 16;

  const v16h a0 = frag_a_h(zb + (size_t)row0 * DHEAD, DHEAD);
  const v16h a1 = frag_a_h(zb + (size_t)row0 * DHEAD + 32, DHEAD);

  for (int nt = 0; nt < DMODEL / 16; ++nt) {
    v8f acc = (v8f){0, 0, 0, 0, 0, 0, 0, 0};
    v16h b0 = frag_b_h(woh + (size_t)(nt * 16) * DHEAD, DHEAD);
    acc = wmma_f32_f16(a0, b0, acc);
    v16h b1 = frag_b_h(woh + (size_t)(nt * 16) * DHEAD + 32, DHEAD);
    acc = wmma_f32_f16(a1, b1, acc);
#pragma unroll
    for (int r = 0; r < 8; ++r) {
      const int grow = row0 + r + 8 * hf;
      out[(size_t)grow * DMODEL + nt * 16 + n] = acc[r];
    }
  }
}

extern "C" void kernel_launch(void* const* d_in, const int* in_sizes, int n_in,
                              void* d_out, int out_size, void* d_ws, size_t ws_size,
                              hipStream_t stream) {
  (void)in_sizes; (void)n_in; (void)out_size; (void)ws_size;
  const float* x  = (const float*)d_in[0];
  // d_in[1] is the causal mask: not read — causality computed from indices.
  const float* Wq = (const float*)d_in[2];
  const float* Wk = (const float*)d_in[3];
  const float* Wv = (const float*)d_in[4];
  const float* Wo = (const float*)d_in[5];
  float* out = (float*)d_out;

  const size_t NX = (size_t)BS * DMODEL;     // 6,291,456
  const size_t NW = (size_t)DHEAD * DMODEL;  // 49,152

  _Float16* xh   = (_Float16*)d_ws;          // (BS, 768) f16
  _Float16* wqh  = xh + NX;                  // (64, 768) f16
  _Float16* wkh  = wqh + NW;
  _Float16* wvh  = wkh + NW;
  _Float16* woh  = wvh + NW;                 // (768, 64) f16
  _Float16* qb   = woh + NW;                 // (BS, 64) f16, pre-scaled
  _Float16* kbuf = qb + (size_t)BS * DHEAD;  // (BS, 64) f16
  _Float16* vT   = kbuf + (size_t)BS * DHEAD;        // (B, 64, SEQ) f16
  _Float16* zb   = vT + (size_t)NBATCH * DHEAD * SEQ;// (BS, 64) f16

  const dim3 blk(128, 1, 1);
  cvt_f16_kernel<<<dim3((unsigned)(NX / 2048)), dim3(256), 0, stream>>>(x, xh, (int)NX);
  cvt_f16_kernel<<<dim3((unsigned)(NW / 2048)), dim3(256), 0, stream>>>(Wq, wqh, (int)NW);
  cvt_f16_kernel<<<dim3((unsigned)(NW / 2048)), dim3(256), 0, stream>>>(Wk, wkh, (int)NW);
  cvt_f16_kernel<<<dim3((unsigned)(NW / 2048)), dim3(256), 0, stream>>>(Wv, wvh, (int)NW);
  cvt_f16_kernel<<<dim3((unsigned)(NW / 2048)), dim3(256), 0, stream>>>(Wo, woh, (int)NW);

  qkv_proj_kernel<<<dim3(BS / 64), blk, 0, stream>>>(xh, wqh, wkh, wvh, qb, kbuf, vT);
  flash_attn_kernel<<<dim3(SEQ / 64, NBATCH), blk, 0, stream>>>(qb, kbuf, vT, zb);
  out_proj_kernel<<<dim3(BS / 64), blk, 0, stream>>>(zb, woh, out);
}
